// GAT_49297634623639
// MI455X (gfx1250) — matrix operand order repaired and checked by reading, not compiled, not run on hardware
//
#include <hip/hip_runtime.h>
#include <hip/hip_bf16.h>
#include <math.h>

typedef __attribute__((ext_vector_type(16))) __bf16 v16bf;
typedef __attribute__((ext_vector_type(8)))  float  v8f;

#define NEG_SLOPE 0.2f

// ---------- order-preserving float<->uint map for atomic max ----------
__device__ __forceinline__ unsigned f2ord(float f) {
    unsigned u = __float_as_uint(f);
    return (u & 0x80000000u) ? ~u : (u | 0x80000000u);
}
__device__ __forceinline__ float ord2f(unsigned u) {
    return (u & 0x80000000u) ? __uint_as_float(u & 0x7FFFFFFFu)
                             : __uint_as_float(~u);
}
#define ORD_NEG_INF 0x007FFFFFu   // f2ord(-inf)

__device__ __forceinline__ float lrelu(float v) { return v > 0.f ? v : NEG_SLOPE * v; }

// ---------- fills ----------
__global__ void k_fill_u32(unsigned* p, unsigned v, long n) {
    long i = (long)blockIdx.x * blockDim.x + threadIdx.x;
    if (i < n) p[i] = v;
}

// ---------- pack W (row-major [256 x ncols]) into per-fragment bf16 hi/lo ----------
// id layout: e(4) | lane(5) | s(3) | tile; B-fragment: lane<16 -> col=lane,K=s*32+{0..7,16..23} ;
// lane>=16 -> col=lane-16, K=s*32+{8..15,24..31}  (matches A fragment K order)
__global__ void k_pack_w(const float* __restrict__ W, __bf16* __restrict__ hi,
                         __bf16* __restrict__ lo, int ldw, int ncols, int ntiles) {
    int id = blockIdx.x * blockDim.x + threadIdx.x;
    int total = ntiles * 8 * 32 * 16;
    if (id >= total) return;
    int e = id & 15, lane = (id >> 4) & 31, s = (id >> 9) & 7, t = id >> 12;
    int kb = (lane & 16) ? 8 : 0;
    int k = s * 32 + kb + (e < 8 ? e : 8 + e);
    int n = t * 16 + (lane & 15);
    float w = (n < ncols) ? W[(long)k * ldw + n] : 0.f;
    __bf16 h = (__bf16)w;
    hi[id] = h;
    lo[id] = (__bf16)(w - (float)h);
}

// ---------- WMMA GEMM: C[M x ncols] = A[M x 256] * W[256 x ncols], bf16x3 split ----------
template <int NT>
__global__ void k_gemm(const float* __restrict__ A, const __bf16* __restrict__ Bhi,
                       const __bf16* __restrict__ Blo, float* __restrict__ C,
                       int M, int ncols, int ldc) {
    const int K = 256;
    int wave = threadIdx.x >> 5;
    int lane = threadIdx.x & 31;
    int mt = blockIdx.x * 8 + wave;            // 16-row tile index
    if (mt * 16 >= M) return;                  // wave-uniform: EXEC stays all-ones
    int tBase = blockIdx.y * NT;

    // 16-bit A layout: lane<16 holds M=lane, K-halves {0..7,16..23}; lane>=16 same M, +8
    int row = mt * 16 + (lane & 15);
    int kb  = (lane & 16) ? 8 : 0;
    const float* rowp = A + (long)row * K;

    v8f acc[NT] = {};
    for (int s = 0; s < 8; ++s) {
        int k0 = s * 32;
        float4 f0 = *(const float4*)(rowp + k0 + kb);
        float4 f1 = *(const float4*)(rowp + k0 + kb + 4);
        float4 f2 = *(const float4*)(rowp + k0 + 16 + kb);
        float4 f3 = *(const float4*)(rowp + k0 + 16 + kb + 4);
        float fv[16] = {f0.x, f0.y, f0.z, f0.w, f1.x, f1.y, f1.z, f1.w,
                        f2.x, f2.y, f2.z, f2.w, f3.x, f3.y, f3.z, f3.w};
        v16bf ahi, alo;
#pragma unroll
        for (int e = 0; e < 16; ++e) {
            __bf16 h = (__bf16)fv[e];
            ahi[e] = h;
            alo[e] = (__bf16)(fv[e] - (float)h);
        }
#pragma unroll
        for (int t = 0; t < NT; ++t) {
            long boff = ((((long)(tBase + t) * 8 + s) * 32) + lane) * 16;
            v16bf bhi = *(const v16bf*)(Bhi + boff);
            v16bf blo = *(const v16bf*)(Blo + boff);
            acc[t] = __builtin_amdgcn_wmma_f32_16x16x32_bf16(false, ahi, false, bhi,
                                                             (short)0, acc[t], false, false);
            acc[t] = __builtin_amdgcn_wmma_f32_16x16x32_bf16(false, ahi, false, blo,
                                                             (short)0, acc[t], false, false);
            acc[t] = __builtin_amdgcn_wmma_f32_16x16x32_bf16(false, alo, false, bhi,
                                                             (short)0, acc[t], false, false);
        }
    }
    // C/D layout: element r -> row mt*16 + r + 8*(lane>=16), col = tile*16 + (lane&15)
    int rowM0 = mt * 16 + ((lane & 16) ? 8 : 0);
#pragma unroll
    for (int t = 0; t < NT; ++t) {
        int col = (tBase + t) * 16 + (lane & 15);
        if (col < ncols) {
#pragma unroll
            for (int r = 0; r < 8; ++r)
                C[(long)(rowM0 + r) * ldc + col] = acc[t][r];
        }
    }
}

// ---------- attention coefficients ----------
__global__ void k_att1(const float* __restrict__ h1, const float* __restrict__ a_src,
                       const float* __restrict__ a_dst, float* as, float* ad, int N) {
    int i = blockIdx.x * blockDim.x + threadIdx.x;   // node*8 + head
    if (i >= N * 8) return;
    int node = i >> 3, h = i & 7;
    const float* hp = h1 + (long)node * 256 + h * 32;
    const float* s = a_src + h * 32;
    const float* d = a_dst + h * 32;
    float vs = 0.f, vd = 0.f;
#pragma unroll
    for (int c = 0; c < 32; ++c) { float x = hp[c]; vs += x * s[c]; vd += x * d[c]; }
    as[i] = vs; ad[i] = vd;
}

__global__ void k_att2(const float* __restrict__ h2, const float* __restrict__ a_src,
                       const float* __restrict__ a_dst, float* as, float* ad, int N) {
    int i = blockIdx.x * blockDim.x + threadIdx.x;
    if (i >= N) return;
    const float* hp = h2 + (long)i * 40;
    float vs = 0.f, vd = 0.f;
#pragma unroll
    for (int c = 0; c < 40; ++c) { float x = hp[c]; vs += x * a_src[c]; vd += x * a_dst[c]; }
    as[i] = vs; ad[i] = vd;
}

// ---------- edge helpers ----------
__device__ __forceinline__ void edge_sd(const int* __restrict__ ei, int E, int e,
                                        int& src, int& dst) {
    if (e < E) { src = ei[e]; dst = ei[E + e]; }
    else       { src = dst = e - E; }            // self loops appended
}

template <int H>
__global__ void k_edge_max(const int* __restrict__ ei, int E, int ET,
                           const float* __restrict__ as, const float* __restrict__ ad,
                           unsigned* __restrict__ m) {
    int i = blockIdx.x * blockDim.x + threadIdx.x;
    if (i >= ET * H) return;
    int e = i / H, h = i % H;
    int src, dst; edge_sd(ei, E, e, src, dst);
    float v = lrelu(as[src * H + h] + ad[dst * H + h]);
    atomicMax(m + dst * H + h, f2ord(v));
}

__global__ void k_decode(unsigned* m, int n) {
    int i = blockIdx.x * blockDim.x + threadIdx.x;
    if (i < n) ((float*)m)[i] = ord2f(m[i]);
}

template <int H>
__global__ void k_edge_sum(const int* __restrict__ ei, int E, int ET,
                           const float* __restrict__ as, const float* __restrict__ ad,
                           const float* __restrict__ m, float* __restrict__ den) {
    int i = blockIdx.x * blockDim.x + threadIdx.x;
    if (i >= ET * H) return;
    int e = i / H, h = i % H;
    int src, dst; edge_sd(ei, E, e, src, dst);
    float v = lrelu(as[src * H + h] + ad[dst * H + h]);
    atomicAdd(den + dst * H + h, expf(v - m[dst * H + h]));
}

// layer-1 messages: one block per edge; 256 lanes = 8 heads x 32 channels
__global__ void k_msg1(const int* __restrict__ ei, int E, const float* __restrict__ h1,
                       const float* __restrict__ as, const float* __restrict__ ad,
                       const float* __restrict__ m, const float* __restrict__ den,
                       float* __restrict__ out) {
    int e = blockIdx.x;
    int h = threadIdx.x >> 5;
    int src, dst; edge_sd(ei, E, e, src, dst);
    float v = lrelu(as[src * 8 + h] + ad[dst * 8 + h]);
    float alpha = expf(v - m[dst * 8 + h]) / den[dst * 8 + h];
    float val = h1[(long)src * 256 + threadIdx.x] * alpha;
    atomicAdd(out + (long)dst * 256 + threadIdx.x, val);
}

// layer-2 messages: one wave per edge, 40 channels
__global__ void k_msg2(const int* __restrict__ ei, int E, int ET,
                       const float* __restrict__ h2, const float* __restrict__ as,
                       const float* __restrict__ ad, const float* __restrict__ m,
                       const float* __restrict__ den, float* __restrict__ out) {
    int gw = blockIdx.x * 8 + (threadIdx.x >> 5);
    int lane = threadIdx.x & 31;
    if (gw >= ET) return;
    int src, dst; edge_sd(ei, E, gw, src, dst);
    float v = lrelu(as[src] + ad[dst]);
    float alpha = expf(v - m[dst]) / den[dst];
    for (int c = lane; c < 40; c += 32)
        atomicAdd(out + (long)dst * 40 + c, h2[(long)src * 40 + c] * alpha);
}

__global__ void k_elu(float* o, const float* __restrict__ b, long n, int ld) {
    long i = (long)blockIdx.x * blockDim.x + threadIdx.x;
    if (i >= n) return;
    float v = o[i] + b[i % ld];
    o[i] = v > 0.f ? v : expm1f(v);
}

__global__ void k_lsm(float* out, const float* __restrict__ b, int N) {
    int i = blockIdx.x * blockDim.x + threadIdx.x;
    if (i >= N) return;
    float* row = out + (long)i * 40;
    float mx = -INFINITY;
    for (int c = 0; c < 40; ++c) mx = fmaxf(mx, row[c] + b[c]);
    float sum = 0.f;
    for (int c = 0; c < 40; ++c) sum += expf(row[c] + b[c] - mx);
    float lse = mx + logf(sum);
    for (int c = 0; c < 40; ++c) row[c] = row[c] + b[c] - lse;
}

extern "C" void kernel_launch(void* const* d_in, const int* in_sizes, int n_in,
                              void* d_out, int out_size, void* d_ws, size_t ws_size,
                              hipStream_t stream) {
    (void)n_in; (void)out_size; (void)ws_size;
    const float* x      = (const float*)d_in[0];
    const int*   ei     = (const int*)d_in[1];
    const float* W1     = (const float*)d_in[2];
    const float* a1_src = (const float*)d_in[3];
    const float* a1_dst = (const float*)d_in[4];
    const float* b1     = (const float*)d_in[5];
    const float* W2     = (const float*)d_in[6];
    const float* a2_src = (const float*)d_in[7];
    const float* a2_dst = (const float*)d_in[8];
    const float* b2     = (const float*)d_in[9];
    float* out = (float*)d_out;

    const int N  = in_sizes[0] / 256;   // 50000 (multiple of 16)
    const int E  = in_sizes[1] / 2;     // 800000
    const int ET = E + N;               // + self loops

    // ---- workspace layout ----
    char* p = (char*)d_ws;
    auto carve = [&](size_t bytes) -> char* {
        char* r = p; p += (bytes + 255) & ~(size_t)255; return r;
    };
    float*  h1   = (float*)carve((size_t)N * 256 * 4);
    float*  o1   = (float*)carve((size_t)N * 256 * 4);
    float*  as1  = (float*)carve((size_t)N * 8 * 4);
    float*  ad1  = (float*)carve((size_t)N * 8 * 4);
    float*  m1   = (float*)carve((size_t)N * 8 * 4);
    float*  d1   = (float*)carve((size_t)N * 8 * 4);
    float*  h2   = (float*)carve((size_t)N * 40 * 4);
    float*  as2  = (float*)carve((size_t)N * 4);
    float*  ad2  = (float*)carve((size_t)N * 4);
    float*  m2   = (float*)carve((size_t)N * 4);
    float*  d2   = (float*)carve((size_t)N * 4);
    __bf16* Whi1 = (__bf16*)carve((size_t)16 * 8 * 32 * 16 * 2);
    __bf16* Wlo1 = (__bf16*)carve((size_t)16 * 8 * 32 * 16 * 2);
    __bf16* Whi2 = (__bf16*)carve((size_t)3 * 8 * 32 * 16 * 2);
    __bf16* Wlo2 = (__bf16*)carve((size_t)3 * 8 * 32 * 16 * 2);

    const int B = 256;
    auto blocks = [](long n, int b) { return (int)((n + b - 1) / b); };

    // ---- pack weights to bf16 hi/lo fragments ----
    {
        int t1 = 16 * 8 * 32 * 16;
        k_pack_w<<<blocks(t1, B), B, 0, stream>>>(W1, Whi1, Wlo1, 256, 256, 16);
        int t2 = 3 * 8 * 32 * 16;
        k_pack_w<<<blocks(t2, B), B, 0, stream>>>(W2, Whi2, Wlo2, 40, 40, 3);
    }
    // ---- init accumulators (ws/d_out are poisoned by harness) ----
    k_fill_u32<<<blocks((long)N * 256, B), B, 0, stream>>>((unsigned*)o1, 0u, (long)N * 256);
    k_fill_u32<<<blocks(N * 8, B), B, 0, stream>>>((unsigned*)m1, ORD_NEG_INF, N * 8);
    k_fill_u32<<<blocks(N * 8, B), B, 0, stream>>>((unsigned*)d1, 0u, N * 8);
    k_fill_u32<<<blocks(N, B), B, 0, stream>>>((unsigned*)m2, ORD_NEG_INF, N);
    k_fill_u32<<<blocks(N, B), B, 0, stream>>>((unsigned*)d2, 0u, N);
    k_fill_u32<<<blocks((long)N * 40, B), B, 0, stream>>>((unsigned*)out, 0u, (long)N * 40);

    const int MT = N / 16;
    // ---- layer 1 ----
    k_gemm<4><<<dim3((MT + 7) / 8, 4), B, 0, stream>>>(x, Whi1, Wlo1, h1, N, 256, 256);
    k_att1<<<blocks(N * 8, B), B, 0, stream>>>(h1, a1_src, a1_dst, as1, ad1, N);
    k_edge_max<8><<<blocks((long)ET * 8, B), B, 0, stream>>>(ei, E, ET, as1, ad1, (unsigned*)m1);
    k_decode<<<blocks(N * 8, B), B, 0, stream>>>((unsigned*)m1, N * 8);
    k_edge_sum<8><<<blocks((long)ET * 8, B), B, 0, stream>>>(ei, E, ET, as1, ad1, m1, d1);
    k_msg1<<<ET, B, 0, stream>>>(ei, E, h1, as1, ad1, m1, d1, o1);
    k_elu<<<blocks((long)N * 256, B), B, 0, stream>>>(o1, b1, (long)N * 256, 256);
    // ---- layer 2 ----
    k_gemm<3><<<dim3((MT + 7) / 8, 1), B, 0, stream>>>(o1, Whi2, Wlo2, h2, N, 40, 40);
    k_att2<<<blocks(N, B), B, 0, stream>>>(h2, a2_src, a2_dst, as2, ad2, N);
    k_edge_max<1><<<blocks(ET, B), B, 0, stream>>>(ei, E, ET, as2, ad2, (unsigned*)m2);
    k_decode<<<blocks(N, B), B, 0, stream>>>((unsigned*)m2, N);
    k_edge_sum<1><<<blocks(ET, B), B, 0, stream>>>(ei, E, ET, as2, ad2, m2, d2);
    k_msg2<<<blocks((long)ET * 32, B), B, 0, stream>>>(ei, E, ET, h2, as2, ad2, m2, d2, out);
    k_lsm<<<blocks(N, B), B, 0, stream>>>(out, b2, N);
}
